// Attention_67611375173689
// MI455X (gfx1250) — compile-verified, matrix operand order checked by reading, not run on hardware
//
#include <hip/hip_runtime.h>
#include <hip/hip_bf16.h>

// ---------------------------------------------------------------------------
// MI455X (gfx1250) SegFormer-style spatial-reduction attention.
// All matmuls in f16 WMMA (v_wmma_f32_16x16x32_f16), fp32 accumulate.
// ---------------------------------------------------------------------------

typedef _Float16 v16h __attribute__((ext_vector_type(16)));
typedef _Float16 v8h  __attribute__((ext_vector_type(8)));
typedef float    v8f  __attribute__((ext_vector_type(8)));

#define BB   4
#define NN   4096     // H*W
#define CC   512
#define NH   8
#define HD   64
#define NKV  1024     // (H/2)*(W/2)
#define KPATCH 2048   // C * SR * SR

__device__ __forceinline__ v8f wmma_f16(v16h a, v16h b, v8f c) {
  return __builtin_amdgcn_wmma_f32_16x16x32_f16(
      /*neg_a=*/false, a, /*neg_b=*/false, b,
      /*c_mod=*/(short)0, c, /*reuse_a=*/false, /*reuse_b=*/false);
}

// Load a 16x32 f16 A/B fragment. `p` points to (row0, k0) of a row-major
// matrix with leading dim `ld` (elements). Per CDNA5 ISA 16-bit layout:
// lane L: row = L%16; VGPR0..3 hold K = (L/16)*8 + 0..7, VGPR4..7 hold
// K = 16 + (L/16)*8 + 0..7  ->  two contiguous 16-byte loads per lane.
__device__ __forceinline__ v16h load_frag(const _Float16* __restrict__ p,
                                          int ld, int lrow, int khalf) {
  const _Float16* q = p + (size_t)lrow * ld + khalf;
  union { v16h v; v8h h[2]; } u;
  u.h[0] = *(const v8h*)(q);
  u.h[1] = *(const v8h*)(q + 16);
  return u.v;
}

// ---------------------------------------------------------------------------
// f32 -> f16 conversion
// ---------------------------------------------------------------------------
__global__ void cvt_f16_kernel(const float* __restrict__ src,
                               _Float16* __restrict__ dst, int n) {
  int i = blockIdx.x * 256 + threadIdx.x;
  if (i < n) dst[i] = (_Float16)src[i];
}

// ---------------------------------------------------------------------------
// im2col for the stride-2 "spatial reduction" conv.
// ap[(b*1024 + ph*32 + pw) * 2048 + ci*4 + i*2 + j] = x[b, (2ph+i)*64+2pw+j, ci]
// sr_w flattened [C][C*4] already matches this K ordering.
// ---------------------------------------------------------------------------
__global__ void build_patches_kernel(const float* __restrict__ x,
                                     _Float16* __restrict__ ap) {
  size_t idx = (size_t)blockIdx.x * 256 + threadIdx.x;
  if (idx >= (size_t)BB * NKV * KPATCH) return;
  int kk = (int)(idx & (KPATCH - 1));
  int p  = (int)(idx >> 11);
  int j  = kk & 1;
  int i  = (kk >> 1) & 1;
  int ci = kk >> 2;
  int pw = p & 31;
  int ph = (p >> 5) & 31;
  int b  = p >> 10;
  int n  = (2 * ph + i) * 64 + 2 * pw + j;
  ap[idx] = (_Float16)x[((size_t)b * NN + n) * CC + ci];
}

// ---------------------------------------------------------------------------
// Generic WMMA GEMM:  D[M, Ntot] = A[M, K] * W[Ntot, K]^T  (+ epilogue)
//   MODE 0: f16 store to out0 (+ optional f32 bias)
//   MODE 1: kv split: cols [0,512) -> f16 kv buffer (k part, row-major);
//           cols [512,1024) -> V transposed into out1 as [b][h][hd][NKV] f16
//   MODE 2: f32 store to out0 + bias
// Block: 256 threads = 8 waves arranged 4x2; wave tile 32x32; block 128x64.
// ---------------------------------------------------------------------------
template <int MODE>
__global__ __launch_bounds__(256) void gemm_wmma_kernel(
    const _Float16* __restrict__ A, const _Float16* __restrict__ W,
    const float* __restrict__ bias, void* __restrict__ out0,
    void* __restrict__ out1, int M, int Ntot, int K) {
  int tid  = threadIdx.x;
  int wave = tid >> 5;
  int lane = tid & 31;
  int wm = wave >> 1;              // 0..3
  int wn = wave & 1;               // 0..1
  int row0 = blockIdx.y * 128 + wm * 32;
  int col0 = blockIdx.x * 64 + wn * 32;
  int lrow  = lane & 15;
  int khalf = (lane >> 4) << 3;    // 0 or 8
  int rbase = khalf;               // same formula for C-layout rows

  v8f acc[2][2] = {};

  for (int k0 = 0; k0 < K; k0 += 32) {
    v16h a0 = load_frag(A + (size_t)(row0 +  0) * K + k0, K, lrow, khalf);
    v16h a1 = load_frag(A + (size_t)(row0 + 16) * K + k0, K, lrow, khalf);
    v16h b0 = load_frag(W + (size_t)(col0 +  0) * K + k0, K, lrow, khalf);
    v16h b1 = load_frag(W + (size_t)(col0 + 16) * K + k0, K, lrow, khalf);
    acc[0][0] = wmma_f16(a0, b0, acc[0][0]);
    acc[0][1] = wmma_f16(a0, b1, acc[0][1]);
    acc[1][0] = wmma_f16(a1, b0, acc[1][0]);
    acc[1][1] = wmma_f16(a1, b1, acc[1][1]);
  }

#pragma unroll
  for (int mt = 0; mt < 2; ++mt) {
#pragma unroll
    for (int nt = 0; nt < 2; ++nt) {
      int gc = col0 + nt * 16 + lrow;
#pragma unroll
      for (int r = 0; r < 8; ++r) {
        int gr = row0 + mt * 16 + rbase + r;
        float v = acc[mt][nt][r];
        if constexpr (MODE == 0) {
          if (bias) v += bias[gc];
          ((_Float16*)out0)[(size_t)gr * Ntot + gc] = (_Float16)v;
        } else if constexpr (MODE == 1) {
          _Float16 hv = (_Float16)v;
          if (gc < CC) {
            ((_Float16*)out0)[(size_t)gr * Ntot + gc] = hv;  // K part
          } else {
            int b = gr >> 10;                  // gr = b*NKV + n
            int n = gr & (NKV - 1);
            int h = (gc - CC) >> 6;
            int d = (gc - CC) & (HD - 1);
            ((_Float16*)out1)[(((size_t)(b * NH + h) * HD + d) << 10) + n] = hv;
          }
        } else {
          ((float*)out0)[(size_t)gr * Ntot + gc] = v + bias[gc];
        }
      }
    }
  }
}

// ---------------------------------------------------------------------------
// Flash attention. Grid: (B*NH, NN/128), 256 threads (8 waves).
// Each wave owns 16 q-rows, streams NKV in chunks of 64 columns (16 WMMAs
// per chunk vs. one softmax reduction pass -> keeps the matrix pipes busy).
//   q  : [B*NN, CC] f16, head slice at col h*HD   (A fragments)
//   kv : [B*NKV, 2*CC] f16, K at cols [0,CC)      (B fragments, contiguous d)
//   vt : [B*NH*HD, NKV] f16 (V transposed)        (B fragments, contiguous n)
// Online softmax with in-wave shfl_xor reductions; P transposed via LDS.
// ---------------------------------------------------------------------------
__global__ __launch_bounds__(256) void attn_kernel(
    const _Float16* __restrict__ qh, const _Float16* __restrict__ kvh,
    const _Float16* __restrict__ vth, _Float16* __restrict__ oh) {
  const float scale = 0.125f;  // hd^-0.5

  int pair = blockIdx.x;       // b*NH + h
  int b = pair >> 3;
  int h = pair & 7;
  int tid  = threadIdx.x;
  int wave = tid >> 5;
  int lane = tid & 31;
  int n0 = blockIdx.y * 128 + wave * 16;   // q row base for this wave
  int lrow  = lane & 15;
  int khalf = (lane >> 4) << 3;
  int rbase = khalf;

  __shared__ __align__(16) _Float16 ptile[8][16 * 64];
  _Float16* myp = ptile[wave];

  const _Float16* qbase = qh + ((size_t)b * NN + n0) * CC + h * HD;
  const _Float16* kbase = kvh + (size_t)b * NKV * (2 * CC) + h * HD;
  const _Float16* vbase = vth + (size_t)pair * HD * NKV;

  v16h qa0 = load_frag(qbase +  0, CC, lrow, khalf);
  v16h qa1 = load_frag(qbase + 32, CC, lrow, khalf);

  v8f oacc[4] = {};
  float mrow[8], srow[8];
#pragma unroll
  for (int r = 0; r < 8; ++r) { mrow[r] = -1e30f; srow[r] = 0.0f; }

  for (int j = 0; j < NKV; j += 64) {
    // ---- S tile (16 x 64) = q @ k^T : 8 WMMAs ----
    v8f sv[4] = {};
#pragma unroll
    for (int nt = 0; nt < 4; ++nt) {
      v16h kb0 = load_frag(kbase + (size_t)(j + nt * 16) * (2 * CC) + 0,
                           2 * CC, lrow, khalf);
      sv[nt] = wmma_f16(qa0, kb0, sv[nt]);
      v16h kb1 = load_frag(kbase + (size_t)(j + nt * 16) * (2 * CC) + 32,
                           2 * CC, lrow, khalf);
      sv[nt] = wmma_f16(qa1, kb1, sv[nt]);
    }
#pragma unroll
    for (int nt = 0; nt < 4; ++nt) sv[nt] = sv[nt] * scale;

    // ---- online softmax (rows live in 16-lane groups; xor masks 1..8) ----
#pragma unroll
    for (int r = 0; r < 8; ++r) {
      float t = fmaxf(fmaxf(sv[0][r], sv[1][r]), fmaxf(sv[2][r], sv[3][r]));
      t = fmaxf(t, __shfl_xor(t, 1, 32));
      t = fmaxf(t, __shfl_xor(t, 2, 32));
      t = fmaxf(t, __shfl_xor(t, 4, 32));
      t = fmaxf(t, __shfl_xor(t, 8, 32));
      float mn   = fmaxf(mrow[r], t);
      float corr = __expf(mrow[r] - mn);
      srow[r] *= corr;
#pragma unroll
      for (int dt = 0; dt < 4; ++dt) oacc[dt][r] *= corr;
      float p0 = __expf(sv[0][r] - mn);
      float p1 = __expf(sv[1][r] - mn);
      float p2 = __expf(sv[2][r] - mn);
      float p3 = __expf(sv[3][r] - mn);
      float ps = (p0 + p1) + (p2 + p3);
      ps += __shfl_xor(ps, 1, 32);
      ps += __shfl_xor(ps, 2, 32);
      ps += __shfl_xor(ps, 4, 32);
      ps += __shfl_xor(ps, 8, 32);
      srow[r] += ps;
      mrow[r] = mn;
      // scatter P (C-layout) into this wave's LDS tile [16 rows][64 cols]
      int prow = rbase + r;
      myp[prow * 64 + lrow]      = (_Float16)p0;
      myp[prow * 64 + 16 + lrow] = (_Float16)p1;
      myp[prow * 64 + 32 + lrow] = (_Float16)p2;
      myp[prow * 64 + 48 + lrow] = (_Float16)p3;
    }
    // intra-wave LDS RAW: DS ops are in-order per wave; wait stores out.
    asm volatile("s_wait_dscnt 0" ::: "memory");

    // reload P as two 16x32 A fragments (layout transpose via LDS)
    v16h pa[2];
#pragma unroll
    for (int kc = 0; kc < 2; ++kc) {
      union { v16h v; v8h h[2]; } u;
      u.h[0] = *(const v8h*)(&myp[lrow * 64 + kc * 32 + khalf]);
      u.h[1] = *(const v8h*)(&myp[lrow * 64 + kc * 32 + 16 + khalf]);
      pa[kc] = u.v;
    }

    // ---- O += P @ V : 8 WMMAs (V pre-transposed: contiguous along n) ----
#pragma unroll
    for (int dt = 0; dt < 4; ++dt) {
#pragma unroll
      for (int kc = 0; kc < 2; ++kc) {
        v16h vb = load_frag(vbase + (size_t)(dt * 16) * NKV + j + kc * 32,
                            NKV, lrow, khalf);
        oacc[dt] = wmma_f16(pa[kc], vb, oacc[dt]);
      }
    }
  }

  // ---- normalize + store (f16, feeds the final projection GEMM) ----
  _Float16* obase = oh + ((size_t)b * NN + n0) * CC + h * HD;
#pragma unroll
  for (int r = 0; r < 8; ++r) {
    float inv = 1.0f / srow[r];
#pragma unroll
    for (int dt = 0; dt < 4; ++dt) {
      obase[(size_t)(rbase + r) * CC + dt * 16 + lrow] =
          (_Float16)(oacc[dt][r] * inv);
    }
  }
}

// ---------------------------------------------------------------------------
// Host-side orchestration
// ---------------------------------------------------------------------------
extern "C" void kernel_launch(void* const* d_in, const int* in_sizes, int n_in,
                              void* d_out, int out_size, void* d_ws,
                              size_t ws_size, hipStream_t stream) {
  const float* x      = (const float*)d_in[0];  // [B, N, C]
  const float* q_w    = (const float*)d_in[1];  // [C, C]
  const float* kv_w   = (const float*)d_in[2];  // [2C, C]
  const float* sr_w   = (const float*)d_in[3];  // [C, C, 2, 2] == [C, 2048]
  const float* sr_b   = (const float*)d_in[4];  // [C]
  const float* proj_w = (const float*)d_in[5];  // [C, C]
  const float* proj_b = (const float*)d_in[6];  // [C]
  float* out = (float*)d_out;                   // [B, N, C] f32

  char* ws = (char*)d_ws;
  size_t off = 0;
  auto alloc = [&](size_t bytes) {
    size_t o = off;
    off += (bytes + 255) & ~(size_t)255;
    return o;
  };
  const size_t XE  = (size_t)BB * NN * CC;        // 8.4M
  const size_t PE  = (size_t)BB * NKV * KPATCH;   // 8.4M
  _Float16* xh      = (_Float16*)(ws + alloc(XE * 2));
  _Float16* patches = (_Float16*)(ws + alloc(PE * 2));
  _Float16* sr_wh   = (_Float16*)(ws + alloc((size_t)CC * KPATCH * 2));
  _Float16* q_wh    = (_Float16*)(ws + alloc((size_t)CC * CC * 2));
  _Float16* kv_wh   = (_Float16*)(ws + alloc((size_t)2 * CC * CC * 2));
  _Float16* proj_wh = (_Float16*)(ws + alloc((size_t)CC * CC * 2));
  _Float16* xconv_h = (_Float16*)(ws + alloc((size_t)BB * NKV * CC * 2));
  _Float16* qh      = (_Float16*)(ws + alloc(XE * 2));
  _Float16* kvh     = (_Float16*)(ws + alloc((size_t)BB * NKV * 2 * CC * 2));
  _Float16* vth     = (_Float16*)(ws + alloc((size_t)BB * NH * HD * NKV * 2));
  _Float16* attnh   = (_Float16*)(ws + alloc(XE * 2));

  auto cvt = [&](const float* s, _Float16* d, size_t n) {
    cvt_f16_kernel<<<dim3((unsigned)((n + 255) / 256)), 256, 0, stream>>>(
        s, d, (int)n);
  };
  cvt(x, xh, XE);
  cvt(q_w, q_wh, (size_t)CC * CC);
  cvt(kv_w, kv_wh, (size_t)2 * CC * CC);
  cvt(sr_w, sr_wh, (size_t)CC * KPATCH);
  cvt(proj_w, proj_wh, (size_t)CC * CC);

  build_patches_kernel<<<dim3((unsigned)((PE + 255) / 256)), 256, 0, stream>>>(
      x, patches);

  // 1) spatial reduction conv as GEMM: [4096, 2048] x [512, 2048]^T + sr_b
  gemm_wmma_kernel<0><<<dim3(CC / 64, (BB * NKV) / 128), 256, 0, stream>>>(
      patches, sr_wh, sr_b, xconv_h, nullptr, BB * NKV, CC, KPATCH);

  // 2) q projection: [16384, 512] x [512, 512]^T
  gemm_wmma_kernel<0><<<dim3(CC / 64, (BB * NN) / 128), 256, 0, stream>>>(
      xh, q_wh, nullptr, qh, nullptr, BB * NN, CC, CC);

  // 3) kv projection: [4096, 512] x [1024, 512]^T, V written transposed
  gemm_wmma_kernel<1><<<dim3((2 * CC) / 64, (BB * NKV) / 128), 256, 0, stream>>>(
      xconv_h, kv_wh, nullptr, kvh, vth, BB * NKV, 2 * CC, CC);

  // 4) flash attention
  attn_kernel<<<dim3(BB * NH, NN / 128), 256, 0, stream>>>(qh, kvh, vth, attnh);

  // 5) output projection: [16384, 512] x [512, 512]^T + proj_b -> f32
  gemm_wmma_kernel<2><<<dim3(CC / 64, (BB * NN) / 128), 256, 0, stream>>>(
      attnh, proj_wh, proj_b, out, nullptr, BB * NN, CC, CC);
}
